// RNNLM_50852412785389
// MI455X (gfx1250) — matrix operand order repaired and checked by reading, not compiled, not run on hardware
//
#include <hip/hip_runtime.h>
#include <cstdint>
#include <cstddef>

// Problem constants (match reference).
#define DMODEL 512
#define SEQ    512
#define BATCH  8
#define VOCAB  32000
#define MROWS  (BATCH * SEQ)   // 4096 rows of [B*T, D]
#define NGATE  (4 * DMODEL)    // 2048
#define KTILES (DMODEL / 32)   // 16 wmma K-steps over K=512

typedef __bf16 v16bf __attribute__((ext_vector_type(16)));
typedef float  v8f   __attribute__((ext_vector_type(8)));

union U32B { uint4 q[2]; v16bf v; };

__device__ __forceinline__ unsigned short f2bf(float f) {
  unsigned u = __float_as_uint(f);
  u += 0x7FFFu + ((u >> 16) & 1u);          // round-to-nearest-even
  return (unsigned short)(u >> 16);
}
__device__ __forceinline__ float sigm(float x) { return 1.0f / (1.0f + __expf(-x)); }

// Async global -> LDS copy of 16B (CDNA5 path, tracked by ASYNCcnt).
__device__ __forceinline__ void async_cp16(unsigned lds_off, const void* gaddr) {
  asm volatile("global_load_async_to_lds_b128 %0, %1, off"
               :: "v"(lds_off), "v"(gaddr)
               : "memory");
}

// ---------------------------------------------------------------------------
// Pack an f32 [K=512, N] row-major weight into the WMMA B-operand layout:
// tile (kt, nt) -> 32 lanes x 16 bf16, lane n<16 holds col nt*16+n, K=kt*32+e;
// lane n>=16 holds col nt*16+(n-16), K=kt*32+16+e. One 32B load/lane at use.
// ---------------------------------------------------------------------------
__global__ void rnnlm_pack_b(const float* __restrict__ W,
                             unsigned short* __restrict__ Pk,
                             int N, int nNt) {
  int tile = blockIdx.x;
  int kt = tile / nNt, nt = tile - kt * nNt;
  int lane = threadIdx.x >> 4;       // 0..31
  int e    = threadIdx.x & 15;       // 0..15
  int krow = kt * 32 + ((lane < 16) ? e : 16 + e);
  int col  = nt * 16 + (lane & 15);
  Pk[(size_t)tile * 512 + lane * 16 + e] = f2bf(W[(size_t)krow * N + col]);
}

// ---------------------------------------------------------------------------
// Embedding gather with ShiftRight(1): x[b,t] = embed[ t==0 ? 0 : tok[b,t-1] ]
// Output row-major bf16 [B*T, D].
// ---------------------------------------------------------------------------
__global__ void rnnlm_embed(const int* __restrict__ tokens,
                            const float* __restrict__ emb,
                            unsigned short* __restrict__ xbf) {
  int row = blockIdx.x;              // b*SEQ + t
  int t = row & (SEQ - 1), b = row >> 9;
  int tok = (t == 0) ? 0 : tokens[b * SEQ + t - 1];
  const float* src = emb + (size_t)tok * DMODEL;
  unsigned short* dst = xbf + (size_t)row * DMODEL;
  for (int d = threadIdx.x; d < DMODEL; d += blockDim.x) dst[d] = f2bf(src[d]);
}

// ---------------------------------------------------------------------------
// Bulk GEMM: C[M,N] = A_bf16[M,512] * Bpk + bias.  8 waves/block; wave owns
// one 16-row M tile and the block's 4 consecutive 16-col N tiles.
// B tiles (4KB per K-step, shared by all 8 waves) are double-buffer staged
// into LDS with async global->LDS copies; waves read them via ds_load_b128.
// grid = (M/128, N/64).
// ---------------------------------------------------------------------------
__global__ void __launch_bounds__(256)
rnnlm_gemm(const unsigned short* __restrict__ A,
           const unsigned short* __restrict__ Bpk,
           const float* __restrict__ bias,
           float* __restrict__ C, int N, int nNt) {
  __shared__ __align__(16) unsigned char bsm[2][4096];

  int tid  = threadIdx.x;
  int wave = tid >> 5;
  int lane = tid & 31;
  int hi = lane >> 4, lr = lane & 15;
  int Mtile = blockIdx.x * 8 + wave;
  int Nt0   = blockIdx.y * 4;

  const unsigned char* bbase = (const unsigned char*)Bpk;

  // Stage issue: 256 threads x 16B = 4KB = the 4 B tiles of one K-step.
  auto issue = [&](int kt, int buf) {
    const unsigned char* g = bbase + (size_t)(kt * nNt + Nt0) * 1024 + tid * 16;
    unsigned l = (unsigned)(size_t)(&bsm[buf][tid * 16]);
    async_cp16(l, (const void*)g);
  };

  const unsigned short* arow = A + (size_t)(Mtile * 16 + lr) * DMODEL;
  v8f zero = {0.f, 0.f, 0.f, 0.f, 0.f, 0.f, 0.f, 0.f};
  v8f acc[4] = {zero, zero, zero, zero};

  issue(0, 0);
  issue(1, 1);

#pragma unroll
  for (int kt = 0; kt < KTILES; ++kt) {
    // Retire stage kt (in-order completion; stage kt+1 may stay in flight).
    if (kt + 1 < KTILES) asm volatile("s_wait_asynccnt 0x1" ::: "memory");
    else                 asm volatile("s_wait_asynccnt 0x0" ::: "memory");
    __syncthreads();                 // stage kt visible to all waves

    U32B a;
    int kb = kt * 32 + hi * 16;      // lanes 0-15: K 0..15; 16-31: K 16..31
    a.q[0] = *(const uint4*)(arow + kb);
    a.q[1] = *(const uint4*)(arow + kb + 8);
#pragma unroll
    for (int nt = 0; nt < 4; ++nt) {
      U32B b;
      const uint4* lp = (const uint4*)&bsm[kt & 1][nt * 1024 + lane * 32];
      b.q[0] = lp[0]; b.q[1] = lp[1];
      acc[nt] = __builtin_amdgcn_wmma_f32_16x16x32_bf16(
          false, a.v, false, b.v, (short)0, acc[nt], false, false);
    }
    __syncthreads();                 // all waves done reading buf[kt&1]
    if (kt + 2 < KTILES) issue(kt + 2, kt & 1);
  }

#pragma unroll
  for (int nt = 0; nt < 4; ++nt) {
    int col = (Nt0 + nt) * 16 + lr;
    float bv = bias[col];
#pragma unroll
    for (int r = 0; r < 8; ++r) {
      int rowg = Mtile * 16 + r + hi * 8;   // C layout: vgpr r -> M = r (+8 hi)
      C[(size_t)rowg * N + col] = acc[nt][r] + bv;
    }
  }
}

// ---------------------------------------------------------------------------
// Device-scope grid barrier (32 blocks, monotonically increasing target).
// ---------------------------------------------------------------------------
__device__ __forceinline__ void grid_barrier(int* bar, int target) {
  __threadfence();
  __syncthreads();
  if (threadIdx.x == 0) {
    __hip_atomic_fetch_add(bar, 1, __ATOMIC_RELEASE, __HIP_MEMORY_SCOPE_AGENT);
    while (__hip_atomic_load(bar, __ATOMIC_ACQUIRE, __HIP_MEMORY_SCOPE_AGENT) < target) {
      __builtin_amdgcn_s_sleep(1);
    }
  }
  __syncthreads();
  __threadfence();
}

// ---------------------------------------------------------------------------
// Persistent LSTM scan. 32 blocks x 128 threads (4 waves). Block w owns
// hidden cols [16w,16w+16); wave g computes gate-g tile (i/f/g/o) of
// gates = h_prev[16x512] * Wh + xW[:,t].  h is double-buffered in global;
// one grid barrier per timestep. Rows 8..15 are zero-padding and stay zero.
// ---------------------------------------------------------------------------
__global__ void __launch_bounds__(128)
rnnlm_scan(const float* __restrict__ xW,           // [4096, 2048] f32 (bias folded)
           const unsigned short* __restrict__ WhPk,// packed B, nNt = 128
           unsigned short* __restrict__ hstate,    // [2][16][512] bf16 (double buf)
           float* __restrict__ cstate,             // [16][512] f32
           unsigned short* __restrict__ hseq,      // [4096, 512] bf16 out
           int* __restrict__ bar) {
  int w    = blockIdx.x;             // hidden slice 0..31
  int gate = threadIdx.x >> 5;       // 0:i 1:f 2:g 3:o
  int lane = threadIdx.x & 31;
  int hi = lane >> 4, lr = lane & 15;
  int nt = gate * 32 + w;            // N-tile in [0,128)
  __shared__ float lds[4][16][16];

  for (int t = 0; t < SEQ; ++t) {
    const unsigned short* hrd = hstate + (size_t)(t & 1) * (16 * DMODEL);
    const unsigned short* arow = hrd + (size_t)lr * DMODEL;
    v8f acc = {0.f, 0.f, 0.f, 0.f, 0.f, 0.f, 0.f, 0.f};
#pragma unroll
    for (int kt = 0; kt < KTILES; ++kt) {
      U32B a, b;
      int kb = kt * 32 + hi * 16;
      a.q[0] = *(const uint4*)(arow + kb);
      a.q[1] = *(const uint4*)(arow + kb + 8);
      const uint4* bp = (const uint4*)(WhPk +
          ((size_t)(kt * 128 + nt) * 512 + lane * 16));
      b.q[0] = bp[0]; b.q[1] = bp[1];
      acc = __builtin_amdgcn_wmma_f32_16x16x32_bf16(
          false, a.v, false, b.v, (short)0, acc, false, false);
    }
    // gates tile -> LDS (add precomputed xW contribution for real batch rows)
#pragma unroll
    for (int r = 0; r < 8; ++r) {
      int m = r + hi * 8;            // padded batch row 0..15
      float v = acc[r];
      if (m < BATCH)
        v += xW[(size_t)(m * SEQ + t) * NGATE + gate * DMODEL + w * 16 + lr];
      lds[gate][m][lr] = v;
    }
    __syncthreads();
    // LSTM pointwise for this block's 16x16 hidden slice
    for (int idx = threadIdx.x; idx < 256; idx += 128) {
      int m = idx >> 4, col = idx & 15;
      float iv = sigm(lds[0][m][col]);
      float fv = sigm(lds[1][m][col]);
      float gv = tanhf(lds[2][m][col]);
      float ov = sigm(lds[3][m][col]);
      int ci = m * DMODEL + w * 16 + col;
      float cn = fv * cstate[ci] + iv * gv;
      cstate[ci] = cn;
      float hn = ov * tanhf(cn);
      unsigned short hb = f2bf(hn);
      hstate[(size_t)((t + 1) & 1) * (16 * DMODEL) + ci] = hb;
      if (m < BATCH)
        hseq[(size_t)(m * SEQ + t) * DMODEL + w * 16 + col] = hb;
    }
    grid_barrier(bar, (t + 1) * (int)gridDim.x);
  }
}

// ---------------------------------------------------------------------------
extern "C" void kernel_launch(void* const* d_in, const int* in_sizes, int n_in,
                              void* d_out, int out_size, void* d_ws, size_t ws_size,
                              hipStream_t stream) {
  const int*   tokens = (const int*)  d_in[0];
  const float* embed  = (const float*)d_in[1];
  const float* Wx0    = (const float*)d_in[2];
  const float* Wh0    = (const float*)d_in[3];
  const float* b0     = (const float*)d_in[4];
  const float* Wx1    = (const float*)d_in[5];
  const float* Wh1    = (const float*)d_in[6];
  const float* b1     = (const float*)d_in[7];
  const float* Wout   = (const float*)d_in[8];
  const float* bout   = (const float*)d_in[9];
  float* logits = (float*)d_out;

  // Workspace carve-up (256B aligned).
  char* ws = (char*)d_ws;
  size_t off = 0;
  auto carve = [&](size_t bytes) -> char* {
    char* p = ws + off;
    off = (off + bytes + 255) & ~(size_t)255;
    return p;
  };
  int*            bar     = (int*)           carve(256);
  unsigned short* xbf     = (unsigned short*)carve((size_t)MROWS * DMODEL * 2);
  unsigned short* h0seq   = (unsigned short*)carve((size_t)MROWS * DMODEL * 2);
  unsigned short* h1seq   = (unsigned short*)carve((size_t)MROWS * DMODEL * 2);
  float*          xWbuf   = (float*)         carve((size_t)MROWS * NGATE * 4);
  unsigned short* Wx0pk   = (unsigned short*)carve((size_t)DMODEL * NGATE * 2);
  unsigned short* Wh0pk   = (unsigned short*)carve((size_t)DMODEL * NGATE * 2);
  unsigned short* Wx1pk   = (unsigned short*)carve((size_t)DMODEL * NGATE * 2);
  unsigned short* Wh1pk   = (unsigned short*)carve((size_t)DMODEL * NGATE * 2);
  unsigned short* Woutpk  = (unsigned short*)carve((size_t)DMODEL * VOCAB * 2);
  unsigned short* hstate  = (unsigned short*)carve((size_t)2 * 16 * DMODEL * 2);
  float*          cstate  = (float*)         carve((size_t)16 * DMODEL * 4);
  (void)in_sizes; (void)n_in; (void)out_size; (void)ws_size;

  const int nNtGate = NGATE / 16;   // 128
  const int nNtOut  = VOCAB / 16;   // 2000

  // 1) Pack weights f32 -> bf16 WMMA-B layout.
  rnnlm_pack_b<<<KTILES * nNtGate, 512, 0, stream>>>(Wx0, Wx0pk, NGATE, nNtGate);
  rnnlm_pack_b<<<KTILES * nNtGate, 512, 0, stream>>>(Wh0, Wh0pk, NGATE, nNtGate);
  rnnlm_pack_b<<<KTILES * nNtGate, 512, 0, stream>>>(Wx1, Wx1pk, NGATE, nNtGate);
  rnnlm_pack_b<<<KTILES * nNtGate, 512, 0, stream>>>(Wh1, Wh1pk, NGATE, nNtGate);
  rnnlm_pack_b<<<KTILES * nNtOut,  512, 0, stream>>>(Wout, Woutpk, VOCAB, nNtOut);

  // 2) Embedding lookup (shifted) -> bf16 activations.
  rnnlm_embed<<<MROWS, 256, 0, stream>>>(tokens, embed, xbf);

  // 3) Layer 0: xW = X*Wx0 + b0 (WMMA), then persistent scan.
  {
    dim3 g(MROWS / 128, NGATE / 64);
    rnnlm_gemm<<<g, 256, 0, stream>>>(xbf, Wx0pk, b0, xWbuf, NGATE, nNtGate);
  }
  hipMemsetAsync(bar, 0, 256, stream);
  hipMemsetAsync(hstate, 0, (size_t)2 * 16 * DMODEL * 2, stream);
  hipMemsetAsync(cstate, 0, (size_t)16 * DMODEL * 4, stream);
  rnnlm_scan<<<32, 128, 0, stream>>>(xWbuf, Wh0pk, hstate, cstate, h0seq, bar);

  // 4) Layer 1.
  {
    dim3 g(MROWS / 128, NGATE / 64);
    rnnlm_gemm<<<g, 256, 0, stream>>>(h0seq, Wx1pk, b1, xWbuf, NGATE, nNtGate);
  }
  hipMemsetAsync(bar, 0, 256, stream);
  hipMemsetAsync(hstate, 0, (size_t)2 * 16 * DMODEL * 2, stream);
  hipMemsetAsync(cstate, 0, (size_t)16 * DMODEL * 4, stream);
  rnnlm_scan<<<32, 128, 0, stream>>>(xWbuf, Wh1pk, hstate, cstate, h1seq, bar);

  // 5) Output projection: logits = H1 * Wout + bout (dominant GEMM, 134 GFLOP).
  {
    dim3 g(MROWS / 128, VOCAB / 64);
    rnnlm_gemm<<<g, 256, 0, stream>>>(h1seq, Woutpk, bout, logits, VOCAB, nNtOut);
  }
}
